// Learned_NodeMLPBPDecoder_14027363188764
// MI455X (gfx1250) — matrix operand order repaired and checked by reading, not compiled, not run on hardware
//
#include <hip/hip_runtime.h>
#include <math.h>

// ---------------------------------------------------------------------------
// Learned node-MLP BP decoder for MI455X (gfx1250, wave32, WMMA bf16).
//
// Sizes (from reference): C=1024 checks, V=2048 vars, DC=6, DV=3, D=8,
// T=3 iterations, B=256 batch. IN_C=48, HC=192, IN_V=25, HV=100.
//
// Strategy: per-node batched GEMMs on v_wmma_f32_16x16x32_bf16.
//   - one wave handles one (node, 16-batch-row tile)
//   - weights repacked once per call into WMMA B-operand lane layout (bf16)
//   - activations kept in bf16 "message buffers" laid out so the WMMA
//     A-operand is two 16B loads per lane (ISA 7.12.2 16-bit A layout)
//   - gelu'd hidden layer staged through wave-private LDS, re-read in A layout
//   - graph gather/scatter fused into GEMM epilogues
//   - gelu uses a branch-free erf rational approx (|err|<=1.5e-7) so the
//     VALU work is small and co-executes with the XDL WMMA pipe
// Entire working set (~142 MB) fits in the 192 MB L2 -> matrix-core bound.
// ---------------------------------------------------------------------------

typedef __attribute__((ext_vector_type(16))) __bf16        v16bf;
typedef __attribute__((ext_vector_type(8)))  float         v8f;
typedef __attribute__((ext_vector_type(4)))  unsigned int  v4u;

static constexpr int NC = 1024;   // checks
static constexpr int NV = 2048;   // variables
static constexpr int NB = 256;    // batch
static constexpr int NT_ITERS = 3;

// padded per-node GEMM dims
static constexpr int KC1 = 64;    // check layer1 K (48 -> 64)
static constexpr int NC1 = 192;   // check layer1 N (=HC)
static constexpr int KC2 = 192;   // check layer2 K
static constexpr int NC2 = 48;    // check layer2 N (3 n-tiles, no pad needed)
static constexpr int KV1 = 32;    // var layer1 K (25 -> 32)
static constexpr int NV1 = 112;   // var layer1 N (100 -> 112)
static constexpr int KV2 = 128;   // var layer2 K (100 -> 128, pads zero)
static constexpr int NV2 = 32;    // var layer2 N (25 -> 32)

// Branch-free erf (Abramowitz & Stegun 7.1.26, |abs err| <= 1.5e-7).
// ~15 VALU ops, no EXEC divergence -> overlaps with WMMA on the XDL pipe.
__device__ __forceinline__ float erf_fast(float x) {
    float ax = __builtin_fabsf(x);
    float t  = __builtin_amdgcn_rcpf(__builtin_fmaf(0.3275911f, ax, 1.0f));
    float p  = 1.061405429f;
    p = __builtin_fmaf(p, t, -1.453152027f);
    p = __builtin_fmaf(p, t, 1.421413741f);
    p = __builtin_fmaf(p, t, -0.284496736f);
    p = __builtin_fmaf(p, t, 0.254829592f);
    p = p * t;
    float e = __expf(-ax * ax);                 // v_exp_f32 based
    float r = __builtin_fmaf(-p, e, 1.0f);
    return __builtin_copysignf(r, x);
}

__device__ __forceinline__ float gelu_exact(float x) {
    return 0.5f * x * (1.0f + erf_fast(x * 0.70710678118654752f));
}

__device__ __forceinline__ v8f wmma_bf16(v16bf a, v16bf b, v8f c) {
    // D = A(16xK=32) * B(K=32 x 16) + C, f32 accumulate
    return __builtin_amdgcn_wmma_f32_16x16x32_bf16(
        false, a, false, b, (short)0, c, false, false);
}

// A-operand load: lane holds row (lane&15); halves at K = k0+8*(lane>>4)
// and K = k0+16+8*(lane>>4), each 8 contiguous bf16 = one 16B load.
__device__ __forceinline__ v16bf load_A(const __bf16* rowp, int k0, int lhi) {
    v16bf a;
    const __bf16* p = rowp + k0 + lhi * 8;
    ((v4u*)&a)[0] = *(const v4u*)(p);
    ((v4u*)&a)[1] = *(const v4u*)(p + 16);
    return a;
}

// ---------------------------------------------------------------------------
// Weight repack: fp32 [node][Kreal][Nreal] -> bf16 WMMA-B tiles.
// Tile (nt,kt) is 512 halves; flat-in-tile = lane*16 + h where lane encodes
// (n_local, (k_local>>3)&1) and h = 8*(k_local>>4) + (k_local&7) -> lane loads 32 contiguous bytes.
// ---------------------------------------------------------------------------
__global__ void repack_weights(const float* __restrict__ src,
                               __bf16* __restrict__ dst,
                               int nodes, int Kreal, int Nreal, int KT, int NT) {
    const int per   = KT * NT * 512;
    const int total = nodes * per;
    for (int e = blockIdx.x * blockDim.x + threadIdx.x; e < total;
         e += gridDim.x * blockDim.x) {
        int node   = e / per;
        int r      = e - node * per;
        int tile   = r >> 9;
        int within = r & 511;
        int nt = tile / KT, kt = tile - nt * KT;
        int lane = within >> 4, h = within & 15;
        int n = nt * 16 + (lane & 15);
        int k = kt * 32 + ((h >> 3) << 4) + ((lane >> 4) << 3) + (h & 7);
        float v = (k < Kreal && n < Nreal)
                      ? src[(node * Kreal + k) * Nreal + n] : 0.0f;
        dst[e] = (__bf16)v;
    }
}

// ---------------------------------------------------------------------------
// chkbuf init: [b][c][64] bf16; k = dc*8+d, d==0 -> prior[c2v_var[c][dc]],
// pads (k>=48) zero.  varbuf init: columns 24..31: col 24 = prior[v], rest 0.
// ---------------------------------------------------------------------------
__global__ void init_chkbuf(__bf16* __restrict__ chkbuf,
                            const float* __restrict__ prior,
                            const int* __restrict__ c2v_var) {
    const int total = NB * NC * 64;
    for (int e = blockIdx.x * blockDim.x + threadIdx.x; e < total;
         e += gridDim.x * blockDim.x) {
        int k = e & 63;
        int c = (e >> 6) & (NC - 1);
        float val = 0.0f;
        if (k < 48 && (k & 7) == 0) val = prior[c2v_var[c * 6 + (k >> 3)]];
        chkbuf[e] = (__bf16)val;
    }
}

__global__ void init_varbuf(__bf16* __restrict__ varbuf,
                            const float* __restrict__ prior) {
    const int total = NB * NV * 8;
    for (int e = blockIdx.x * blockDim.x + threadIdx.x; e < total;
         e += gridDim.x * blockDim.x) {
        int k8 = e & 7;
        int v  = (e >> 3) & (NV - 1);
        int b  = e >> 14;
        varbuf[(b * NV + v) * 32 + 24 + k8] = (__bf16)(k8 == 0 ? prior[v] : 0.0f);
    }
}

// ---------------------------------------------------------------------------
// Check-node MLP. One wave per (check c, batch tile of 16 rows).
//   X[16x64] -> H[16x192] (gelu, via LDS) -> Y[16x48] * sgn -> scatter varbuf
// ---------------------------------------------------------------------------
__global__ __launch_bounds__(256) void check_mlp(
    const __bf16* __restrict__ chkbuf, const __bf16* __restrict__ Wc1b,
    const float* __restrict__ bc1,     const __bf16* __restrict__ Wc2b,
    const float* __restrict__ bc2,     const int* __restrict__ synd,
    const int* __restrict__ c2v_var,   const int* __restrict__ c2v_pos,
    __bf16* __restrict__ varbuf) {
    __shared__ __bf16 lds[8 * 16 * NC1];             // 48 KB: 6 KB per wave
    const int lane  = threadIdx.x & 31;
    const int wave  = threadIdx.x >> 5;
    const int nl    = lane & 15;
    const int lhi   = lane >> 4;
    const int c     = blockIdx.x;
    const int btile = blockIdx.y * 8 + wave;
    __bf16* H = lds + wave * (16 * NC1);

    // pull the layer-2 weight block toward the WGP while layer 1 runs
    const __bf16* w2 = Wc2b + c * (KC2 * NC2);       // 9216 halves
#pragma unroll
    for (int q = 0; q < 4; ++q)
        __builtin_prefetch(w2 + (q * 32 + lane) * 144, 0, 1);

    // syndrome signs for this wave's 8 accumulator rows (per C/D layout:
    // row m = i + 8*(lane>>4), n = lane&15)
    float sg[8];
#pragma unroll
    for (int i = 0; i < 8; ++i) {
        int brow = btile * 16 + lhi * 8 + i;
        sg[i] = 1.0f - 2.0f * (float)synd[brow * NC + c];
    }

    // ---- layer 1: [16x64] x [64x192] ----
    const __bf16* arow = chkbuf + ((btile * 16 + nl) * NC + c) * KC1;
    v16bf a0 = load_A(arow, 0, lhi);
    v16bf a1 = load_A(arow, 32, lhi);
    const __bf16* w1 = Wc1b + c * (KC1 * NC1);
#pragma unroll
    for (int nt = 0; nt < NC1 / 16; ++nt) {
        v16bf b0 = *(const v16bf*)(w1 + (nt * 2 + 0) * 512 + lane * 16);
        v16bf b1 = *(const v16bf*)(w1 + (nt * 2 + 1) * 512 + lane * 16);
        v8f acc = {};
        acc = wmma_bf16(a0, b0, acc);
        acc = wmma_bf16(a1, b1, acc);
        float bias = bc1[c * NC1 + nt * 16 + nl];
#pragma unroll
        for (int i = 0; i < 8; ++i)
            H[(lhi * 8 + i) * NC1 + nt * 16 + nl] =
                (__bf16)gelu_exact(acc[i] + bias);
    }

    // ---- layer 2: [16x192] x [192x48], sign + graph scatter epilogue ----
#pragma unroll
    for (int nt = 0; nt < NC2 / 16; ++nt) {
        v8f acc = {};
#pragma unroll
        for (int kt = 0; kt < KC2 / 32; ++kt) {
            v16bf a;
            const __bf16* hrow = H + nl * NC1 + kt * 32 + lhi * 8;
            ((v4u*)&a)[0] = *(const v4u*)(hrow);
            ((v4u*)&a)[1] = *(const v4u*)(hrow + 16);
            v16bf b = *(const v16bf*)(w2 + (nt * 6 + kt) * 512 + lane * 16);
            acc = wmma_bf16(a, b, acc);
        }
        int   n    = nt * 16 + nl;              // output feature 0..47
        float bias = bc2[c * NC2 + n];
        int   dc = n >> 3, d = n & 7;
        int   j = c2v_var[c * 6 + dc];
        int   p = c2v_pos[c * 6 + dc];
#pragma unroll
        for (int i = 0; i < 8; ++i) {
            int brow = btile * 16 + lhi * 8 + i;
            varbuf[(brow * NV + j) * 32 + p * 8 + d] =
                (__bf16)((acc[i] + bias) * sg[i]);
        }
    }
}

// ---------------------------------------------------------------------------
// Variable-node MLP. One wave per (variable v, batch tile of 16 rows).
//   X[16x32] -> H[16x112->128] (gelu, via LDS) -> Y[16x32]
//   cols 0..23 scatter to chkbuf, col 24 = LLR -> d_out[t].
// ---------------------------------------------------------------------------
__global__ __launch_bounds__(256) void var_mlp(
    const __bf16* __restrict__ varbuf, const __bf16* __restrict__ Wv1b,
    const float* __restrict__ bv1,     const __bf16* __restrict__ Wv2b,
    const float* __restrict__ bv2,     const int* __restrict__ v2c_chk,
    const int* __restrict__ v2c_pos,   __bf16* __restrict__ chkbuf,
    float* __restrict__ out, int t) {
    __shared__ __bf16 lds[8 * 16 * KV2];             // 32 KB: 4 KB per wave
    const int lane  = threadIdx.x & 31;
    const int wave  = threadIdx.x >> 5;
    const int nl    = lane & 15;
    const int lhi   = lane >> 4;
    const int v     = blockIdx.x;
    const int btile = blockIdx.y * 8 + wave;
    __bf16* H = lds + wave * (16 * KV2);

    // prefetch layer-2 weights (4096 halves) while layer 1 runs
    const __bf16* w2 = Wv2b + v * (KV2 * NV2);
    __builtin_prefetch(w2 + lane * 128, 0, 1);

    // zero K-pad columns 112..127 of H
#pragma unroll
    for (int i = 0; i < 8; ++i)
        H[(lhi * 8 + i) * KV2 + 112 + nl] = (__bf16)0.0f;

    // ---- layer 1: [16x32] x [32x112] ----
    const __bf16* arow = varbuf + ((btile * 16 + nl) * NV + v) * KV1;
    v16bf a0 = load_A(arow, 0, lhi);
    const __bf16* w1 = Wv1b + v * (KV1 * NV1);
#pragma unroll
    for (int nt = 0; nt < NV1 / 16; ++nt) {
        v16bf b = *(const v16bf*)(w1 + nt * 512 + lane * 16);
        v8f acc = {};
        acc = wmma_bf16(a0, b, acc);
        int   n    = nt * 16 + nl;
        float bias = (n < 100) ? bv1[v * 100 + n] : 0.0f;   // pad rows -> gelu(0)=0
#pragma unroll
        for (int i = 0; i < 8; ++i)
            H[(lhi * 8 + i) * KV2 + n] = (__bf16)gelu_exact(acc[i] + bias);
    }

    // ---- layer 2: [16x128] x [128x32], scatter + LLR epilogue ----
#pragma unroll
    for (int nt = 0; nt < NV2 / 16; ++nt) {
        v8f acc = {};
#pragma unroll
        for (int kt = 0; kt < KV2 / 32; ++kt) {
            v16bf a;
            const __bf16* hrow = H + nl * KV2 + kt * 32 + lhi * 8;
            ((v4u*)&a)[0] = *(const v4u*)(hrow);
            ((v4u*)&a)[1] = *(const v4u*)(hrow + 16);
            v16bf b = *(const v16bf*)(w2 + (nt * 4 + kt) * 512 + lane * 16);
            acc = wmma_bf16(a, b, acc);
        }
        int n = nt * 16 + nl;                   // output feature 0..31
        if (n < 24) {                           // messages back to checks
            int   p  = n >> 3, d = n & 7;
            int   ic = v2c_chk[v * 3 + p];
            int   kp = v2c_pos[v * 3 + p];
            float bias = bv2[v * 25 + n];
#pragma unroll
            for (int i = 0; i < 8; ++i) {
                int brow = btile * 16 + lhi * 8 + i;
                chkbuf[(brow * NC + ic) * 64 + kp * 8 + d] =
                    (__bf16)(acc[i] + bias);
            }
        } else if (n == 24) {                   // LLR output (fp32)
            float bias = bv2[v * 25 + 24];
#pragma unroll
            for (int i = 0; i < 8; ++i) {
                int brow = btile * 16 + lhi * 8 + i;
                out[t * (NB * NV) + brow * NV + v] = acc[i] + bias;
            }
        }
    }
}

// ---------------------------------------------------------------------------
extern "C" void kernel_launch(void* const* d_in, const int* in_sizes, int n_in,
                              void* d_out, int out_size, void* d_ws,
                              size_t ws_size, hipStream_t stream) {
    const int*   synd    = (const int*)d_in[0];
    const float* prior   = (const float*)d_in[1];
    const float* Wc1     = (const float*)d_in[2];
    const float* bc1     = (const float*)d_in[3];
    const float* Wc2     = (const float*)d_in[4];
    const float* bc2     = (const float*)d_in[5];
    const float* Wv1     = (const float*)d_in[6];
    const float* bv1     = (const float*)d_in[7];
    const float* Wv2     = (const float*)d_in[8];
    const float* bv2     = (const float*)d_in[9];
    const int*   c2v_var = (const int*)d_in[10];
    const int*   c2v_pos = (const int*)d_in[11];
    const int*   v2c_chk = (const int*)d_in[12];
    const int*   v2c_pos = (const int*)d_in[13];
    float*       out     = (float*)d_out;

    char*  ws  = (char*)d_ws;
    size_t off = 0;
    auto take = [&](size_t bytes) {
        void* p = ws + off;
        off = (off + bytes + 255) & ~(size_t)255;
        return p;
    };
    __bf16* wc1b   = (__bf16*)take((size_t)NC * KC1 * NC1 * 2); // 25.2 MB
    __bf16* wc2b   = (__bf16*)take((size_t)NC * KC2 * NC2 * 2); // 18.9 MB
    __bf16* wv1b   = (__bf16*)take((size_t)NV * KV1 * NV1 * 2); // 14.7 MB
    __bf16* wv2b   = (__bf16*)take((size_t)NV * KV2 * NV2 * 2); // 16.8 MB
    __bf16* chkbuf = (__bf16*)take((size_t)NB * NC * 64 * 2);   // 33.6 MB
    __bf16* varbuf = (__bf16*)take((size_t)NB * NV * 32 * 2);   // 33.6 MB

    // one-time (per call) weight repack fp32 -> WMMA-B bf16 tiles
    repack_weights<<<4096, 256, 0, stream>>>(Wc1, wc1b, NC, 48, 192, 2, 12);
    repack_weights<<<4096, 256, 0, stream>>>(Wc2, wc2b, NC, 192, 48, 6, 3);
    repack_weights<<<4096, 256, 0, stream>>>(Wv1, wv1b, NV, 25, 100, 1, 7);
    repack_weights<<<4096, 256, 0, stream>>>(Wv2, wv2b, NV, 100, 25, 4, 2);

    init_chkbuf<<<4096, 256, 0, stream>>>(chkbuf, prior, c2v_var);
    init_varbuf<<<2048, 256, 0, stream>>>(varbuf, prior);

    dim3 blk(256);
    dim3 grid_c(NC, NB / 16 / 8);   // (1024, 2), 8 waves/block
    dim3 grid_v(NV, NB / 16 / 8);   // (2048, 2)
    for (int t = 0; t < NT_ITERS; ++t) {
        check_mlp<<<grid_c, blk, 0, stream>>>(chkbuf, wc1b, bc1, wc2b, bc2,
                                              synd, c2v_var, c2v_pos, varbuf);
        var_mlp<<<grid_v, blk, 0, stream>>>(varbuf, wv1b, bv1, wv2b, bv2,
                                            v2c_chk, v2c_pos, chkbuf, out, t);
    }
}
